// AttnBlock_1726576856113
// MI455X (gfx1250) — matrix operand order repaired and checked by reading, compiled, not run on hardware
//
#include <hip/hip_runtime.h>
#include <hip/hip_bf16.h>

#define B_   4
#define C_   512
#define N_   4096
#define G_   32
#define CPG  16
#define SCALE 0.044194173824159216f   // 512^-0.5

typedef __bf16 bf16_t;
typedef bf16_t v16bf __attribute__((ext_vector_type(16)));
typedef float  v8f   __attribute__((ext_vector_type(8)));
typedef float  f32x4 __attribute__((ext_vector_type(4)));
typedef unsigned short u16x8 __attribute__((ext_vector_type(8)));

union V16 { v16bf v; unsigned short u[16]; };

__device__ __forceinline__ unsigned short f2bf(float f) {
  union { float f; unsigned u; } x; x.f = f;
  unsigned r = x.u + 0x7FFFu + ((x.u >> 16) & 1u);
  return (unsigned short)(r >> 16);
}
// Load 8 contiguous bf16 (16 bytes) into fragment slots [lo, lo+8)
__device__ __forceinline__ void load8g(V16& f, int lo, const unsigned short* p) {
  u16x8 t = *(const u16x8*)p;
#pragma unroll
  for (int e = 0; e < 8; ++e) f.u[lo + e] = t[e];
}
// A fragment (16x32): elements 0..15 <- K = half*16+0..15
__device__ __forceinline__ void loadA(V16& f, const unsigned short* row, int k0, int half) {
  load8g(f, 0, row + k0 + half * 16);
  load8g(f, 8, row + k0 + half * 16 + 8);
}
// B fragment (32x16) from K-contiguous storage: elements 0..15 <- K = half*16+0..15
__device__ __forceinline__ void loadB(V16& f, const unsigned short* col, int k0, int half) {
  load8g(f, 0, col + k0 + half * 16);
  load8g(f, 8, col + k0 + half * 16 + 8);
}

// ---------------- GroupNorm ----------------
__global__ __launch_bounds__(256) void gn_stats(const float* __restrict__ x,
                                                float* __restrict__ stats) {
  int bg = blockIdx.x;                  // b*32+g
  const float* p = x + (size_t)bg * CPG * N_;
  int tid = threadIdx.x;
  float s = 0.f, ss = 0.f;
  for (int i = tid; i < CPG * N_; i += 256) { float v = p[i]; s += v; ss += v * v; }
  __shared__ float shs[256], shq[256];
  shs[tid] = s; shq[tid] = ss;
  __syncthreads();
  for (int off = 128; off > 0; off >>= 1) {
    if (tid < off) { shs[tid] += shs[tid + off]; shq[tid] += shq[tid + off]; }
    __syncthreads();
  }
  if (tid == 0) {
    float inv = 1.0f / (CPG * N_);
    float mean = shs[0] * inv;
    float var  = shq[0] * inv - mean * mean;
    stats[bg * 2]     = mean;
    stats[bg * 2 + 1] = rsqrtf(var + 1e-6f);
  }
}

// tiled normalize + transpose: x[b][c][n] -> h_t[b][n][c] (bf16), coalesced both sides
__global__ __launch_bounds__(256) void gn_apply(const float* __restrict__ x,
                                                const float* __restrict__ gamma,
                                                const float* __restrict__ beta,
                                                const float* __restrict__ stats,
                                                unsigned short* __restrict__ h_t) {
  int n0 = blockIdx.x * 32, c0 = blockIdx.y * 32, b = blockIdx.z;
  __shared__ float tile[32][33];
  int tx = threadIdx.x & 31, ty = threadIdx.x >> 5;   // 32 x 8
#pragma unroll
  for (int rr = ty; rr < 32; rr += 8) {
    int c = c0 + rr;
    float mean = stats[(b * G_ + c / CPG) * 2];
    float rstd = stats[(b * G_ + c / CPG) * 2 + 1];
    tile[rr][tx] = (x[((size_t)b * C_ + c) * N_ + n0 + tx] - mean) * rstd * gamma[c] + beta[c];
  }
  __syncthreads();
#pragma unroll
  for (int rr = ty; rr < 32; rr += 8)
    h_t[((size_t)b * N_ + n0 + rr) * C_ + c0 + tx] = f2bf(tile[tx][rr]);
}

// ---------------- weight convert (f32 -> bf16) ----------------
__global__ __launch_bounds__(256) void wconv(const float* __restrict__ qw,
                                             const float* __restrict__ kw,
                                             const float* __restrict__ vw,
                                             const float* __restrict__ pw,
                                             unsigned short* __restrict__ wqkv,
                                             unsigned short* __restrict__ wproj) {
  int i = blockIdx.x * 256 + threadIdx.x;  // 0..C*C-1
  wqkv[i]               = f2bf(qw[i]);
  wqkv[C_ * C_ + i]     = f2bf(kw[i]);
  wqkv[2 * C_ * C_ + i] = f2bf(vw[i]);
  wproj[i]              = f2bf(pw[i]);
}

// ---------------- fused QKV GEMM: [1536x512] x [512x4096] per batch -------------
// h_t is [b][n][c]; outputs: qT,kT as [b][n][c] (q pre-scaled), v as [b][c][n].
__global__ __launch_bounds__(256) void qkv_gemm(const unsigned short* __restrict__ W,
                                                const unsigned short* __restrict__ h_t,
                                                const float* __restrict__ qb,
                                                const float* __restrict__ kb,
                                                const float* __restrict__ vb,
                                                unsigned short* __restrict__ qT,
                                                unsigned short* __restrict__ kT,
                                                unsigned short* __restrict__ vo) {
  int nt = blockIdx.x;                 // 64 tiles of N=64
  int mt = blockIdx.y;                 // 12 tiles of M=128 (sel uniform per block)
  int b  = blockIdx.z;
  int wave = threadIdx.x >> 5, lane = threadIdx.x & 31;
  int half = lane >> 4, ln = lane & 15;
  int m0 = mt * 128 + wave * 16;
  int n0 = nt * 64;
  const size_t bN = (size_t)b * N_;
  const unsigned short* Arow = W + (size_t)(m0 + ln) * C_;

  __shared__ __align__(16) unsigned short VT[128 * 64];   // v transpose stage

  v8f acc[4] = {};
  for (int k0 = 0; k0 < C_; k0 += 32) {
    V16 a;
    loadA(a, Arow, k0, half);
    if (k0 + 32 < C_) __builtin_prefetch(Arow + k0 + 32, 0, 1);
#pragma unroll
    for (int j = 0; j < 4; ++j) {
      V16 bb;
      loadB(bb, h_t + (bN + n0 + j * 16 + ln) * C_, k0, half);
      acc[j] = __builtin_amdgcn_wmma_f32_16x16x32_bf16(false, a.v, false, bb.v,
                                                       (short)0, acc[j], false, false);
    }
  }
  int sel  = m0 >> 9;          // 0=q 1=k 2=v (uniform per BLOCK: mt*128 spans one region)
  int mrow = m0 & (C_ - 1);
  if (sel == 0) {              // q: transposed + scale, 16B stores
#pragma unroll
    for (int j = 0; j < 4; ++j) {
      u16x8 st;
#pragma unroll
      for (int r = 0; r < 8; ++r) st[r] = f2bf((acc[j][r] + qb[mrow + r + half * 8]) * SCALE);
      *(u16x8*)(qT + (bN + n0 + j * 16 + ln) * C_ + mrow + half * 8) = st;
    }
  } else if (sel == 1) {       // k: transposed, 16B stores
#pragma unroll
    for (int j = 0; j < 4; ++j) {
      u16x8 st;
#pragma unroll
      for (int r = 0; r < 8; ++r) st[r] = f2bf(acc[j][r] + kb[mrow + r + half * 8]);
      *(u16x8*)(kT + (bN + n0 + j * 16 + ln) * C_ + mrow + half * 8) = st;
    }
  } else {                     // v [c][n]: stage transpose in LDS
#pragma unroll
    for (int j = 0; j < 4; ++j)
#pragma unroll
      for (int r = 0; r < 8; ++r) {
        int row = wave * 16 + r + half * 8;              // 0..127 within tile
        VT[row * 64 + j * 16 + ln] = f2bf(acc[j][r] + vb[mrow + r + half * 8]);
      }
  }
  __syncthreads();
  if (sel == 2) {
    int vr0 = mrow & ~127;                               // tile base row in v
    unsigned short* out = vo + (size_t)b * C_ * N_;
#pragma unroll
    for (int ch = 0; ch < 4; ++ch) {
      int idx = (threadIdx.x + ch * 256) * 8;            // 8192 u16 total
      int row = idx >> 6, col = idx & 63;
      *(u16x8*)(out + (size_t)(vr0 + row) * N_ + n0 + col) = *(const u16x8*)(&VT[idx]);
    }
  }
}

// ---------------- flash attention over N=4096 per batch ----------------
// grid: B*(N/16) blocks, 128 threads (4 waves). Each block: 16 queries, full C.
// wave w owns output channel slice [128w, 128w+128).  qT/kT are [n][c], v is [c][n].
__global__ __launch_bounds__(128) void attn_kernel(const unsigned short* __restrict__ qT,
                                                   const unsigned short* __restrict__ kT,
                                                   const unsigned short* __restrict__ v,
                                                   unsigned short* __restrict__ aoT) {
  int blk = blockIdx.x;
  int b = blk >> 8;
  int i0 = (blk & 255) * 16;
  const size_t bN = (size_t)b * N_;
  const unsigned short* qp = qT + bN * C_;
  const unsigned short* kp = kT + bN * C_;
  const unsigned short* vp = v + bN * C_;   // [c][n]

  __shared__ __align__(16) unsigned short qs[16 * 512];  // q tile [i][c]; reused as O^T stage
  __shared__ __align__(16) float S[16][64];
  __shared__ __align__(16) unsigned short P[16][64];
  __shared__ float mrun[16], lrun[16], alpha_s[16];

  int tid = threadIdx.x;
  int wave = tid >> 5, lane = tid & 31, half = lane >> 4, ln = lane & 15;

  // ---- async-copy contiguous 16KB q tile (rows i0..i0+15 of qT) into LDS
  {
    const unsigned short* g0 = qp + (size_t)i0 * C_;
    unsigned lds0 = (unsigned)(unsigned long long)(&qs[0]);
#pragma unroll
    for (int ch = 0; ch < 8; ++ch) {
      int idx = (tid + ch * 128) * 8;                    // 8 u16 per 16B chunk
      unsigned lds_addr = lds0 + (unsigned)(idx * 2);
      unsigned long long ga = (unsigned long long)(g0 + idx);
      asm volatile("global_load_async_to_lds_b128 %0, %1, off"
                   :: "v"(lds_addr), "v"(ga) : "memory");
    }
    asm volatile("s_wait_asynccnt 0x0" ::: "memory");
  }
  if (tid < 16) { mrun[tid] = -1e30f; lrun[tid] = 0.f; }
  __syncthreads();

  v8f O[8] = {};
  int c0 = wave * 128;
  const unsigned short* qrow = &qs[ln * 512];            // A rows from LDS

  for (int j0 = 0; j0 < N_; j0 += 64) {
    // ---- S tile: wave computes 16x16 scores for columns jw..jw+15, K=512
    v8f sacc = {};
    int jw = j0 + wave * 16;
    const unsigned short* krow = kp + (size_t)(jw + ln) * C_;
    for (int kc = 0; kc < C_; kc += 32) {
      V16 a, bb;
      loadA(a, qrow, kc, half);                          // ds_load_b128 x2
      loadB(bb, krow, kc, half);                         // global_load_b128 x2
      sacc = __builtin_amdgcn_wmma_f32_16x16x32_bf16(false, a.v, false, bb.v,
                                                     (short)0, sacc, false, false);
    }
#pragma unroll
    for (int r = 0; r < 8; ++r) S[r + half * 8][wave * 16 + ln] = sacc[r];
    __syncthreads();

    // ---- online softmax row update (one thread per query row)
    if (tid < 16) {
      float m_old = mrun[tid];
      float mx = m_old;
      for (int j = 0; j < 64; ++j) mx = fmaxf(mx, S[tid][j]);
      float al = __expf(m_old - mx);
      float ls = lrun[tid] * al;
      for (int j = 0; j < 64; ++j) {
        float p = __expf(S[tid][j] - mx);
        ls += p;
        P[tid][j] = f2bf(p);
      }
      mrun[tid] = mx; lrun[tid] = ls; alpha_s[tid] = al;
    }
    __syncthreads();

    // ---- rescale running O, then O += P[16x64] @ v^T[64 x 128-slice]
#pragma unroll
    for (int t = 0; t < 8; ++t)
#pragma unroll
      for (int r = 0; r < 8; ++r) O[t][r] *= alpha_s[r + half * 8];

#pragma unroll
    for (int kk = 0; kk < 64; kk += 32) {
      V16 a;
      loadA(a, &P[ln][0], kk, half);                     // ds_load_b128 x2
#pragma unroll
      for (int t = 0; t < 8; ++t) {
        V16 bb;
        load8g(bb, 0, vp + (size_t)(c0 + t * 16 + ln) * N_ + j0 + kk + half * 16);
        load8g(bb, 8, vp + (size_t)(c0 + t * 16 + ln) * N_ + j0 + kk + half * 16 + 8);
        O[t] = __builtin_amdgcn_wmma_f32_16x16x32_bf16(false, a.v, false, bb.v,
                                                       (short)0, O[t], false, false);
      }
    }
  }

  // ---- epilogue: O / l -> LDS transpose (reuse qs) -> 16B coalesced stores
  unsigned short* OT = qs;                               // [16][512]
#pragma unroll
  for (int t = 0; t < 8; ++t)
#pragma unroll
    for (int r = 0; r < 8; ++r)
      OT[(r + half * 8) * 512 + c0 + t * 16 + ln] = f2bf(O[t][r] / lrun[r + half * 8]);
  __syncthreads();
  unsigned short* aop = aoT + (bN + i0) * C_;            // tile is contiguous 16KB
#pragma unroll
  for (int ch = 0; ch < 8; ++ch) {
    int idx = (tid + ch * 128) * 8;
    *(u16x8*)(aop + idx) = *(const u16x8*)(&OT[idx]);
  }
}

// ---------------- proj GEMM + bias + residual (f32 out) ----------------
// aoT is [b][n][c]; output is [b][c][n] f32, staged via LDS for coalesced f32x4 stores.
__global__ __launch_bounds__(256) void proj_gemm(const unsigned short* __restrict__ W,
                                                 const unsigned short* __restrict__ aoT,
                                                 const float* __restrict__ pb,
                                                 const float* __restrict__ x,
                                                 float* __restrict__ out) {
  int nt = blockIdx.x;                 // 64
  int mt = blockIdx.y;                 // 4 (M=512/128)
  int b  = blockIdx.z;
  int wave = threadIdx.x >> 5, lane = threadIdx.x & 31;
  int half = lane >> 4, ln = lane & 15;
  int m0 = mt * 128 + wave * 16;
  int n0 = nt * 64;
  const size_t bN = (size_t)b * N_;
  const unsigned short* Arow = W + (size_t)(m0 + ln) * C_;

  __shared__ __align__(16) float PT[128 * 64];           // 32KB f32 transpose stage

  v8f acc[4] = {};
  for (int k0 = 0; k0 < C_; k0 += 32) {
    V16 a;
    loadA(a, Arow, k0, half);
    if (k0 + 32 < C_) __builtin_prefetch(Arow + k0 + 32, 0, 1);
#pragma unroll
    for (int j = 0; j < 4; ++j) {
      V16 bb;
      loadB(bb, aoT + (bN + n0 + j * 16 + ln) * C_, k0, half);
      acc[j] = __builtin_amdgcn_wmma_f32_16x16x32_bf16(false, a.v, false, bb.v,
                                                       (short)0, acc[j], false, false);
    }
  }
#pragma unroll
  for (int j = 0; j < 4; ++j)
#pragma unroll
    for (int r = 0; r < 8; ++r) {
      int row = wave * 16 + r + half * 8;                // 0..127 in tile
      PT[row * 64 + j * 16 + ln] = acc[j][r] + pb[m0 + r + half * 8];
    }
  __syncthreads();
  const float* xb = x + (size_t)b * C_ * N_;
  float* ob = out + (size_t)b * C_ * N_;
  int r0 = mt * 128;
#pragma unroll
  for (int ch = 0; ch < 8; ++ch) {
    int idx = (threadIdx.x + ch * 256) * 4;              // 8192 f32 total
    int row = idx >> 6, col = idx & 63;
    size_t off = (size_t)(r0 + row) * N_ + n0 + col;
    f32x4 xv = *(const f32x4*)(xb + off);
    f32x4 pv = *(const f32x4*)(&PT[idx]);
    *(f32x4*)(ob + off) = pv + xv;
  }
}

extern "C" void kernel_launch(void* const* d_in, const int* in_sizes, int n_in,
                              void* d_out, int out_size, void* d_ws, size_t ws_size,
                              hipStream_t stream) {
  const float* x      = (const float*)d_in[0];
  const float* norm_w = (const float*)d_in[1];
  const float* norm_b = (const float*)d_in[2];
  const float* q_w    = (const float*)d_in[3];
  const float* q_b    = (const float*)d_in[4];
  const float* k_w    = (const float*)d_in[5];
  const float* k_b    = (const float*)d_in[6];
  const float* v_w    = (const float*)d_in[7];
  const float* v_b    = (const float*)d_in[8];
  const float* p_w    = (const float*)d_in[9];
  const float* p_b    = (const float*)d_in[10];
  float* out = (float*)d_out;

  char* ws = (char*)d_ws;
  const size_t SZ_BCN = (size_t)B_ * C_ * N_ * sizeof(unsigned short); // 16 MB
  float*          stats = (float*)ws;          ws += 1024;
  unsigned short* h_t   = (unsigned short*)ws; ws += SZ_BCN;
  unsigned short* wqkv  = (unsigned short*)ws; ws += 3 * C_ * C_ * sizeof(unsigned short);
  unsigned short* wproj = (unsigned short*)ws; ws += C_ * C_ * sizeof(unsigned short);
  unsigned short* qTb   = (unsigned short*)ws; ws += SZ_BCN;
  unsigned short* kTb   = (unsigned short*)ws; ws += SZ_BCN;
  unsigned short* vbuf  = (unsigned short*)ws; ws += SZ_BCN;
  unsigned short* aoT   = (unsigned short*)ws; ws += SZ_BCN;

  gn_stats <<<B_ * G_, 256, 0, stream>>>(x, stats);
  gn_apply <<<dim3(N_ / 32, C_ / 32, B_), 256, 0, stream>>>(x, norm_w, norm_b, stats, h_t);
  wconv    <<<(C_ * C_) / 256, 256, 0, stream>>>(q_w, k_w, v_w, p_w, wqkv, wproj);
  qkv_gemm <<<dim3(N_ / 64, (3 * C_) / 128, B_), 256, 0, stream>>>(wqkv, h_t,
                                                                   q_b, k_b, v_b,
                                                                   qTb, kTb, vbuf);
  attn_kernel<<<B_ * (N_ / 16), 128, 0, stream>>>(qTb, kTb, vbuf, aoT);
  proj_gemm<<<dim3(N_ / 64, C_ / 128, B_), 256, 0, stream>>>(wproj, aoT, p_b, x, out);
}
